// GATBioNet_66907000537560
// MI455X (gfx1250) — matrix operand order repaired and checked by reading, not compiled, not run on hardware
//
#include <hip/hip_runtime.h>

// ---------------------------------------------------------------------------
// GAT (2-layer) for MI455X / gfx1250.  Dense linear layers use
// v_wmma_f32_16x16x32_f16 (wave32 WMMA); edge softmax/aggregation is
// warp-per-edge with global_atomic_add_f32 (L2-resident: 192MB L2 holds the
// whole graph working set).
// ---------------------------------------------------------------------------

typedef __attribute__((ext_vector_type(16))) _Float16 v16h;
typedef __attribute__((ext_vector_type(8)))  float    v8f;

static constexpr int FIN = 256;  // layer-1 input dim
static constexpr int H1  = 4;    // heads, layer 1
static constexpr int C1  = 32;   // per-head channels, layer 1
static constexpr int F1  = 128;  // H1*C1
static constexpr int F2  = 64;   // layer-2 output dim (1 head)
static constexpr float NEG_SLOPE = 0.2f;

// ---------------------------------------------------------------------------
__global__ void k_zero_f32(float* __restrict__ p, long n) {
  long i = (long)blockIdx.x * blockDim.x + threadIdx.x;
  if (i < n) p[i] = 0.0f;
}

__global__ void k_f32_to_f16(const float* __restrict__ s, _Float16* __restrict__ d, long n) {
  long i = (long)blockIdx.x * blockDim.x + threadIdx.x;
  if (i < n) d[i] = (_Float16)s[i];
}

// W [K x Ncol] row-major (f32)  ->  Wt [Ncol x K] row-major (f16)
__global__ void k_transpose_to_f16(const float* __restrict__ W, _Float16* __restrict__ Wt,
                                   int K, int Ncol) {
  int i = blockIdx.x * blockDim.x + threadIdx.x;
  if (i >= K * Ncol) return;
  int n = i / K, k = i % K;
  Wt[(size_t)n * K + k] = (_Float16)W[(size_t)k * Ncol + n];
}

// ---------------------------------------------------------------------------
// C[M,N] = A[M,K] (f16, row-major) x Bt[N,K] (f16, N-major == B transposed)
// One wave per 16x16 tile, blockDim = (32, N/16).  K consumed 32/step via
// v_wmma_f32_16x16x32_f16.  M, K multiples of 16/32 (50000, 256/128 here).
//
// VGPR layouts used here (A/B k order is contiguous, not the ISA table order):
//   A 16x32 f16 : lane m=lane&15 ; halves 0..15 -> K = kk + (lane>=16?16:0)+i
//   B 32x16 f16 : lane n=lane&15 ; halves 0..15 -> K = kk + (lane>=16?16:0)+i
//   C/D 16x16 f32: vgpr r -> row r + (lane>=16?8:0), col lane&15
// ---------------------------------------------------------------------------
template <int K>
__global__ void k_wmma_gemm(const _Float16* __restrict__ A,
                            const _Float16* __restrict__ Bt,
                            float* __restrict__ C, int M, int N) {
  const int lane   = threadIdx.x;        // 0..31
  const int half16 = (lane >= 16);
  const int m0     = blockIdx.x * 16;
  const int n0     = threadIdx.y * 16;
  const int mrow   = m0 + (lane & 15);
  const int ncol   = n0 + (lane & 15);

  v8f acc = {0.f, 0.f, 0.f, 0.f, 0.f, 0.f, 0.f, 0.f};

  const uint4* Arow = reinterpret_cast<const uint4*>(A + (size_t)mrow * K);
  const uint4* Brow = reinterpret_cast<const uint4*>(Bt + (size_t)ncol * K);

#pragma unroll
  for (int kk = 0; kk < K; kk += 32) {
    union { v16h v; uint4 u[2]; } a, b;
    const int ka = (kk + (half16 ? 16 : 0)) >> 3;   // uint4 (8-half) units
    a.u[0] = Arow[ka];          // halves 0..7  : K = base+0..7
    a.u[1] = Arow[ka + 1];      // halves 8..15 : K = base+8..15
    const int kb = (kk + (half16 ? 16 : 0)) >> 3;
    b.u[0] = Brow[kb];          // halves 0..7  : K = base+0..7
    b.u[1] = Brow[kb + 1];      // halves 8..15 : K = base+8..15
    acc = __builtin_amdgcn_wmma_f32_16x16x32_f16(
        /*neg_a=*/false, a.v, /*neg_b=*/false, b.v,
        /*c_mod=*/(short)0, acc, /*reuse_a=*/false, /*reuse_b=*/false);
  }

  const int mo = m0 + (half16 ? 8 : 0);
#pragma unroll
  for (int r = 0; r < 8; ++r)
    C[(size_t)(mo + r) * N + ncol] = acc[r];
}

// ---------------------------------------------------------------------------
// aS[n,h] = sum_c Hm[n, h*Cc+c] * att_s[h*Cc+c]  (one wave per node)
// ---------------------------------------------------------------------------
__global__ void k_att_coef(const float* __restrict__ Hm,
                           const float* __restrict__ att_s,
                           const float* __restrict__ att_d,
                           float* __restrict__ aS, float* __restrict__ aD,
                           int Nn, int Hh, int Cc) {
  int wave = (int)(((long)blockIdx.x * blockDim.x + threadIdx.x) >> 5);
  int lane = threadIdx.x & 31;
  if (wave >= Nn) return;
  const float* row = Hm + (size_t)wave * Hh * Cc;
  for (int h = 0; h < Hh; ++h) {
    float s = 0.f, d = 0.f;
    for (int c = lane; c < Cc; c += 32) {
      float v = row[h * Cc + c];
      s += v * att_s[h * Cc + c];
      d += v * att_d[h * Cc + c];
    }
#pragma unroll
    for (int off = 16; off > 0; off >>= 1) {
      s += __shfl_down(s, off, 32);
      d += __shfl_down(d, off, 32);
    }
    if (lane == 0) { aS[(size_t)wave * Hh + h] = s; aD[(size_t)wave * Hh + h] = d; }
  }
}

// ---------------------------------------------------------------------------
// Per edge: e = exp(leaky_relu(aS[src]+aD[dst]));  eexp[e,h] = e;
// denom[dst,h] += e.   (softmax max-shift skipped: logits are O(1))
// ---------------------------------------------------------------------------
__global__ void k_edge_exp(const int* __restrict__ src, const int* __restrict__ dst,
                           const float* __restrict__ aS, const float* __restrict__ aD,
                           float* __restrict__ eexp, float* __restrict__ denom,
                           int E, int Hh) {
  int e = blockIdx.x * blockDim.x + threadIdx.x;
  if (e >= E) return;
  int s = src[e], d = dst[e];
  for (int h = 0; h < Hh; ++h) {
    float x = aS[(size_t)s * Hh + h] + aD[(size_t)d * Hh + h];
    x = (x > 0.f) ? x : NEG_SLOPE * x;
    float ex = __expf(x);
    eexp[(size_t)e * Hh + h] = ex;
    atomicAdd(&denom[(size_t)d * Hh + h], ex);
  }
}

// ---------------------------------------------------------------------------
// Warp-per-edge message scatter: out[dst, ch] += Hm[src, ch] * alpha(head(ch))
// Channels F = Hh*Cc covered 32 lanes at a time (coalesced gather + f32 atomics)
// ---------------------------------------------------------------------------
__global__ void k_scatter(const int* __restrict__ src, const int* __restrict__ dst,
                          const float* __restrict__ Hm,
                          const float* __restrict__ eexp, const float* __restrict__ denom,
                          float* __restrict__ out, int E, int Hh, int Cc) {
  int e = (int)(((long)blockIdx.x * blockDim.x + threadIdx.x) >> 5);
  int lane = threadIdx.x & 31;
  if (e >= E) return;
  int s = src[e], d = dst[e];
  const int F = Hh * Cc;
  for (int j = 0; j < F / 32; ++j) {
    int ch = j * 32 + lane;
    int h  = ch / Cc;
    float alpha = eexp[(size_t)e * Hh + h] / (denom[(size_t)d * Hh + h] + 1e-16f);
    atomicAdd(&out[(size_t)d * F + ch], Hm[(size_t)s * F + ch] * alpha);
  }
}

__global__ void k_bias_relu_f16(const float* __restrict__ acc, const float* __restrict__ b,
                                _Float16* __restrict__ oh, long n, int F) {
  long i = (long)blockIdx.x * blockDim.x + threadIdx.x;
  if (i >= n) return;
  float v = acc[i] + b[i % F];
  oh[i] = (_Float16)(v > 0.f ? v : 0.f);
}

__global__ void k_bias_add(float* __restrict__ out, const float* __restrict__ b,
                           long n, int F) {
  long i = (long)blockIdx.x * blockDim.x + threadIdx.x;
  if (i < n) out[i] += b[i % F];
}

// ---------------------------------------------------------------------------
extern "C" void kernel_launch(void* const* d_in, const int* in_sizes, int n_in,
                              void* d_out, int out_size, void* d_ws, size_t ws_size,
                              hipStream_t stream) {
  (void)n_in; (void)out_size; (void)ws_size;
  const float* X   = (const float*)d_in[0];
  const int*   ei  = (const int*)d_in[1];       // [2, E] (src row, dst row)
  const float* W1  = (const float*)d_in[2];
  const float* as1 = (const float*)d_in[3];
  const float* ad1 = (const float*)d_in[4];
  const float* b1  = (const float*)d_in[5];
  const float* W2  = (const float*)d_in[6];
  const float* as2 = (const float*)d_in[7];
  const float* ad2 = (const float*)d_in[8];
  const float* b2  = (const float*)d_in[9];
  float* out = (float*)d_out;

  const int N = in_sizes[0] / FIN;   // 50000 (multiple of 16)
  const int E = in_sizes[1] / 2;     // 800000
  const int* esrc = ei;
  const int* edst = ei + E;

  // ---- workspace carve-out (256B aligned) ----
  char* w = (char*)d_ws;
  auto alloc = [&](size_t bytes) -> void* {
    void* p = (void*)w; w += (bytes + 255) & ~(size_t)255; return p;
  };
  _Float16* Xh  = (_Float16*)alloc((size_t)N * FIN * 2); // X in f16
  _Float16* W1t = (_Float16*)alloc((size_t)F1 * FIN * 2);
  _Float16* W2t = (_Float16*)alloc((size_t)F2 * F1 * 2);
  float*    h1  = (float*)alloc((size_t)N * F1 * 4);     // X@W1
  float*    aS1 = (float*)alloc((size_t)N * H1 * 4);
  float*    aD1 = (float*)alloc((size_t)N * H1 * 4);
  float*    ex1 = (float*)alloc((size_t)E * H1 * 4);     // exp(e) per edge/head
  float*    dn1 = (float*)alloc((size_t)N * H1 * 4);     // softmax denominators
  float*    o1  = (float*)alloc((size_t)N * F1 * 4);     // aggregated layer-1 out
  _Float16* h1a = (_Float16*)alloc((size_t)N * F1 * 2);  // relu(o1+b1) in f16
  float*    h2  = (float*)alloc((size_t)N * F2 * 4);     // h1a@W2
  float*    aS2 = (float*)alloc((size_t)N * 4);
  float*    aD2 = (float*)alloc((size_t)N * 4);
  float*    ex2 = (float*)alloc((size_t)E * 4);
  float*    dn2 = (float*)alloc((size_t)N * 4);

  auto nb = [](long n, int t) { return (unsigned)((n + t - 1) / t); };

  // ---- precision prep ----
  k_f32_to_f16<<<nb((long)N * FIN, 256), 256, 0, stream>>>(X, Xh, (long)N * FIN);
  k_transpose_to_f16<<<nb(FIN * F1, 256), 256, 0, stream>>>(W1, W1t, FIN, F1);
  k_transpose_to_f16<<<nb(F1 * F2, 256), 256, 0, stream>>>(W2, W2t, F1, F2);

  // ---- zero accumulators (ws and d_out are poisoned by the harness) ----
  k_zero_f32<<<nb((long)N * H1, 256), 256, 0, stream>>>(dn1, (long)N * H1);
  k_zero_f32<<<nb((long)N * F1, 256), 256, 0, stream>>>(o1, (long)N * F1);
  k_zero_f32<<<nb(N, 256), 256, 0, stream>>>(dn2, N);
  k_zero_f32<<<nb((long)N * F2, 256), 256, 0, stream>>>(out, (long)N * F2);

  // ---- layer 1 ----
  k_wmma_gemm<FIN><<<dim3(N / 16), dim3(32, F1 / 16), 0, stream>>>(Xh, W1t, h1, N, F1);
  k_att_coef<<<nb((long)N * 32, 256), 256, 0, stream>>>(h1, as1, ad1, aS1, aD1, N, H1, C1);
  k_edge_exp<<<nb(E, 256), 256, 0, stream>>>(esrc, edst, aS1, aD1, ex1, dn1, E, H1);
  k_scatter<<<nb((long)E * 32, 256), 256, 0, stream>>>(esrc, edst, h1, ex1, dn1, o1, E, H1, C1);
  k_bias_relu_f16<<<nb((long)N * F1, 256), 256, 0, stream>>>(o1, b1, h1a, (long)N * F1, F1);

  // ---- layer 2 ----
  k_wmma_gemm<F1><<<dim3(N / 16), dim3(32, F2 / 16), 0, stream>>>(h1a, W2t, h2, N, F2);
  k_att_coef<<<nb((long)N * 32, 256), 256, 0, stream>>>(h2, as2, ad2, aS2, aD2, N, 1, F2);
  k_edge_exp<<<nb(E, 256), 256, 0, stream>>>(esrc, edst, aS2, aD2, ex2, dn2, E, 1);
  k_scatter<<<nb((long)E * 32, 256), 256, 0, stream>>>(esrc, edst, h2, ex2, dn2, out, E, 1, F2);
  k_bias_add<<<nb((long)N * F2, 256), 256, 0, stream>>>(out, b2, (long)N * F2, F2);
}
